// ConfidenceDecoder_8409545965733
// MI455X (gfx1250) — matrix operand order repaired and checked by reading, not compiled, not run on hardware
//
#include <hip/hip_runtime.h>
#include <hip/hip_bf16.h>
#include <math.h>

typedef __attribute__((ext_vector_type(16))) _Float16 v16h;
typedef __attribute__((ext_vector_type(4)))  _Float16 v4h;
typedef __attribute__((ext_vector_type(8)))  float    v8f;

union Frag16 { v16h v; _Float16 h[16]; };
union Acc8   { v8f  v; float    f[8]; };

__device__ __forceinline__ int iclamp(int x, int hi) { return x < hi ? x : hi; }

// ---------------------------------------------------------------------------
// Generic tiled WMMA GEMM: C[M,N] = A[M,K] @ B[K,N] (+bias) (+=C if accum)
// fp32 global <-> f16 LDS staging, fp32 accumulate via v_wmma_f32_16x16x32_f16.
// Block = 128 threads (4 waves), 32x32 C tile, each wave one 16x16 WMMA tile.
// GUARD=false fast path: M,N,K multiples of 32 -> unguarded float4 loads,
// packed b64 LDS stores for the A tile, unconditional epilogue stores.
// GUARD=true: clamped (always in-bounds) scalar loads + value select; no
// exec-mask branches around any load.
// ---------------------------------------------------------------------------
template <bool GUARD>
__global__ __launch_bounds__(128) void wmma_gemm_kernel(
    const float* __restrict__ A, const float* __restrict__ B,
    const float* __restrict__ bias, float* __restrict__ C,
    int M, int N, int K, int lda, int ldb, int ldc, int accum)
{
    __shared__ _Float16 lA[32][36];   // [m][k]  (pitch 36: b64-aligned packed stores)
    __shared__ _Float16 lB[32][36];   // transposed: [n][k] so (k,k+1) pairs are contiguous
    const int tid  = threadIdx.x;
    const int lane = tid & 31;
    const int wave = tid >> 5;
    const int wm   = (wave >> 1) * 16;
    const int wn   = (wave & 1) * 16;
    const int bm   = blockIdx.x * 32;
    const int bn   = blockIdx.y * 32;
    const int l15  = lane & 15;
    const int hlf  = lane >> 4;

    Acc8 acc;
    #pragma unroll
    for (int r = 0; r < 8; ++r) acc.f[r] = 0.0f;

    for (int k0 = 0; k0 < K; k0 += 32) {
        if (!GUARD) {
            // 32x32 tile = 256 float4 loads for each of A and B; 2 per thread.
            #pragma unroll
            for (int i = 0; i < 2; ++i) {
                int idx = tid + i * 128;
                int r  = idx >> 3;
                int c4 = (idx & 7) * 4;
                float4 av = *(const float4*)(A + (long)(bm + r) * lda + (k0 + c4));
                v4h ap = { (_Float16)av.x, (_Float16)av.y, (_Float16)av.z, (_Float16)av.w };
                *(v4h*)(&lA[r][c4]) = ap;                 // single ds_store_b64
                float4 bv = *(const float4*)(B + (long)(k0 + r) * ldb + (bn + c4));
                lB[c4 + 0][r] = (_Float16)bv.x;
                lB[c4 + 1][r] = (_Float16)bv.y;
                lB[c4 + 2][r] = (_Float16)bv.z;
                lB[c4 + 3][r] = (_Float16)bv.w;
            }
        } else {
            #pragma unroll
            for (int i = 0; i < 8; ++i) {
                int idx = tid + i * 128;
                int r = idx >> 5, c = idx & 31;
                int gr = bm + r, gc = k0 + c;
                float av = A[(long)iclamp(gr, M - 1) * lda + iclamp(gc, K - 1)];
                av *= (gr < M && gc < K) ? 1.0f : 0.0f;   // arithmetic blend: no sink
                lA[r][c] = (_Float16)av;
                int gk = k0 + r, gn = bn + c;
                float bv = B[(long)iclamp(gk, K - 1) * ldb + iclamp(gn, N - 1)];
                bv *= (gk < K && gn < N) ? 1.0f : 0.0f;
                lB[c][r] = (_Float16)bv;
            }
        }
        __syncthreads();
        Frag16 af, bf;
        #pragma unroll
        for (int t = 0; t < 8; ++t) {
            // A 16x32 f16 layout: lane m=l15, half selects K block of 8
            int ka = (t >> 2) * 16 + hlf * 8 + (t & 3) * 2;
            af.h[2*t]   = lA[wm + l15][ka];
            af.h[2*t+1] = lA[wm + l15][ka + 1];
            // B 32x16 f16 layout: lane n=l15, half selects K block of 8
            int kb = (t >> 2) * 16 + hlf * 8 + (t & 3) * 2;
            bf.h[2*t]   = lB[wn + l15][kb];
            bf.h[2*t+1] = lB[wn + l15][kb + 1];
        }
        acc.v = __builtin_amdgcn_wmma_f32_16x16x32_f16(false, af.v, false, bf.v,
                                                       (short)0, acc.v, false, false);
        __syncthreads();
    }

    // Branch-free epilogue loads (bias/accum are wave-uniform); fast path
    // stores unconditionally.
    const int col  = bn + wn + l15;
    const int colc = iclamp(col, N - 1);
    float bval = 0.0f;
    if (bias) bval = bias[colc];
    #pragma unroll
    for (int r = 0; r < 8; ++r) {
        int row  = bm + wm + r + 8 * hlf;
        int rowc = iclamp(row, M - 1);
        float o = acc.f[r] + bval;
        if (accum) o += C[(long)rowc * ldc + colc];
        if (!GUARD) {
            C[(long)row * ldc + col] = o;
        } else if (row < M && col < N) {
            C[(long)row * ldc + col] = o;
        }
    }
}

// ---------------------------------------------------------------------------
// Fused attention per (b,h): scores = scale * A .* (q k^T), mask, softmax,
// ctx = P @ v. WMMA for both GEMMs; softmax in LDS fp32.
// grid = BS*NH = 64 blocks, 256 threads (8 waves).
// S pitch 113; cols 100..112 zeroed after softmax -> phase 3 is guard-free.
// Out-of-tile score lanes write to a dump slot -> stores unconditional, no
// branch for the compiler to sink the adj/Amat loads into.
// ---------------------------------------------------------------------------
#define SP_  113
#define DUMP_ (100 * SP_)

__global__ __launch_bounds__(256) void attn_fused_kernel(
    const float* __restrict__ q, const float* __restrict__ k, const float* __restrict__ v,
    const float* __restrict__ Amat, const int* __restrict__ adj,
    const int* __restrict__ lens, float* __restrict__ ctx, int mode)
{
    __shared__ float    S[100 * SP_ + 1];  // + dump slot
    __shared__ _Float16 Vt[64 * 130];      // V transposed: [d][j], j padded to 128 with zeros
    const int b    = blockIdx.x >> 3;
    const int h    = blockIdx.x & 7;
    const int tid  = threadIdx.x;
    const int lane = tid & 31;
    const int wave = tid >> 5;
    const int l15  = lane & 15;
    const int hlf  = lane >> 4;
    const int len  = lens[b];
    const float scale = 0.125f;            // 1/sqrt(64)
    const float* qh = q + (long)b * 100 * 512 + h * 64;
    const float* kh = k + (long)b * 100 * 512 + h * 64;
    const float* vh = v + (long)b * 100 * 512 + h * 64;

    // Stage V^T into LDS (j >= 100 zero-padded up to 128; unguarded B-frags later)
    for (int i = tid; i < 64 * 128; i += 256) {
        int d = i >> 7, j = i & 127;
        float vv = vh[(long)iclamp(j, 99) * 512 + d];
        vv *= (j < 100) ? 1.0f : 0.0f;     // arithmetic blend: load stays unconditional
        Vt[d * 130 + j] = (_Float16)vv;
    }

    // Phase 1: scores (7x7 tiles of 16x16, K=64 in two WMMA steps).
    // No zero-selects on q/k: garbage rows/cols >=100 go to the dump slot.
    const long adjb = (long)b * 10000;
    const long amb  = ((long)b * 8 + h) * 10000;
    for (int t = wave; t < 49; t += 8) {
        const int ti = (t / 7) * 16;
        const int tj = (t % 7) * 16;
        Acc8 acc;
        #pragma unroll
        for (int r = 0; r < 8; ++r) acc.f[r] = 0.0f;
        const float* qrow = qh + (long)iclamp(ti + l15, 99) * 512;
        const float* krow = kh + (long)iclamp(tj + l15, 99) * 512;
        for (int kk = 0; kk < 64; kk += 32) {
            Frag16 af, bf;
            #pragma unroll
            for (int t8 = 0; t8 < 8; ++t8) {
                int ka = kk + (t8 >> 2) * 16 + hlf * 8 + (t8 & 3) * 2;
                float2 qq = *(const float2*)(qrow + ka);
                af.h[2*t8]   = (_Float16)qq.x;
                af.h[2*t8+1] = (_Float16)qq.y;
                int kb = kk + (t8 >> 2) * 16 + hlf * 8 + (t8 & 3) * 2;
                float2 kq = *(const float2*)(krow + kb);
                bf.h[2*t8]   = (_Float16)kq.x;
                bf.h[2*t8+1] = (_Float16)kq.y;
            }
            acc.v = __builtin_amdgcn_wmma_f32_16x16x32_f16(false, af.v, false, bf.v,
                                                           (short)0, acc.v, false, false);
        }
        // Batched unconditional adj/Amat loads, then arithmetic mask blend and
        // unconditional LDS stores (dump slot for out-of-tile lanes).
        int   avv[8];
        float Avv[8];
        #pragma unroll
        for (int r = 0; r < 8; ++r) {
            int i  = ti + r + 8 * hlf;
            int j  = tj + l15;
            int ic = iclamp(i, 99);
            int jc = iclamp(j, 99);
            avv[r] = adj[adjb + (long)ic * 100 + jc];
            Avv[r] = Amat[amb + (long)ic * 100 + jc];
        }
        #pragma unroll
        for (int r = 0; r < 8; ++r) {
            int i = ti + r + 8 * hlf;
            int j = tj + l15;
            bool hit = (mode == 0) ? (avv[r] == 0) : (avv[r] != 0);
            bool masked = (hit || (j >= len)) && (i != j);   // eye forced unmasked
            float mf = masked ? 1.0f : 0.0f;
            float sv = (1.0f - mf) * (acc.f[r] * scale * Avv[r]) + mf * (-1e30f);
            int addr = (i < 100 && j < 100) ? (i * SP_ + j) : DUMP_;
            S[addr] = sv;
        }
    }
    __syncthreads();

    // Phase 2: softmax, one row per thread; zero the K-padding columns.
    if (tid < 100) {
        float mx = -3.4e38f;
        for (int j = 0; j < 100; ++j) mx = fmaxf(mx, S[tid * SP_ + j]);
        float sum = 0.0f;
        for (int j = 0; j < 100; ++j) {
            float e = expf(S[tid * SP_ + j] - mx);
            S[tid * SP_ + j] = e;
            sum += e;
        }
        float inv = 1.0f / sum;
        for (int j = 0; j < 100; ++j) S[tid * SP_ + j] *= inv;
        for (int j = 100; j < SP_; ++j) S[tid * SP_ + j] = 0.0f;
    }
    __syncthreads();

    // Phase 3: ctx = P @ V  (7 x 4 tiles, K padded to 128) -- fully guard-free
    // fragment reads: S cols 100..112 are zero, Vt cols 100..127 are zero,
    // clamped rows only affect unsaved output rows.
    for (int t = wave; t < 28; t += 8) {
        const int ti = (t >> 2) * 16;
        const int td = (t & 3) * 16;
        Acc8 acc;
        #pragma unroll
        for (int r = 0; r < 8; ++r) acc.f[r] = 0.0f;
        const int pic = iclamp(ti + l15, 99);
        const int vd  = td + l15;
        for (int j0 = 0; j0 < 112; j0 += 32) {
            Frag16 af, bf;
            #pragma unroll
            for (int t8 = 0; t8 < 8; ++t8) {
                int ja = j0 + (t8 >> 2) * 16 + hlf * 8 + (t8 & 3) * 2;
                af.h[2*t8]   = (_Float16)S[pic * SP_ + ja];
                af.h[2*t8+1] = (_Float16)S[pic * SP_ + ja + 1];
                int jb = j0 + (t8 >> 2) * 16 + hlf * 8 + (t8 & 3) * 2;          // < 128: zero-padded in Vt
                bf.h[2*t8]   = Vt[vd * 130 + jb];
                bf.h[2*t8+1] = Vt[vd * 130 + jb + 1];
            }
            acc.v = __builtin_amdgcn_wmma_f32_16x16x32_f16(false, af.v, false, bf.v,
                                                           (short)0, acc.v, false, false);
        }
        #pragma unroll
        for (int r = 0; r < 8; ++r) {
            int i = ti + r + 8 * hlf;
            if (i < 100) ctx[((long)b * 100 + i) * 512 + h * 64 + vd] = acc.f[r];
        }
    }
}

// ---------------------------------------------------------------------------
// Edge generator: A[b,h,i,j] = sigmoid( sum_c relu(pa+pb+pc+b1)[c] * w2[c,h] + b2[h] )
// One block per (b,i); caches pa-row+b1 and w2 in LDS. float4 streaming loads.
// ---------------------------------------------------------------------------
__global__ __launch_bounds__(128) void edge_gen_kernel(
    const float* __restrict__ pa, const float* __restrict__ pb, const float* __restrict__ pc,
    const float* __restrict__ b1, const float* __restrict__ w2, const float* __restrict__ b2,
    float* __restrict__ Aout)
{
    __shared__ float base[512];
    __shared__ float w2l[512 * 8];
    const int blk = blockIdx.x;
    const int b = blk / 100, i = blk % 100;
    const int tid = threadIdx.x;
    for (int c = tid; c < 512;  c += 128) base[c] = pa[((long)b * 100 + i) * 512 + c] + b1[c];
    for (int c = tid; c < 4096; c += 128) w2l[c]  = w2[c];
    __syncthreads();
    for (int j = tid; j < 100; j += 128) {
        float acc[8];
        #pragma unroll
        for (int hh = 0; hh < 8; ++hh) acc[hh] = 0.0f;
        const float* pbr = pb + ((long)b * 100 + j) * 512;
        const float* pcr = pc + (long)(j - i + 100) * 512;
        for (int c = 0; c < 512; c += 4) {
            float4 pbv = *(const float4*)(pbr + c);
            float4 pcv = *(const float4*)(pcr + c);
            float hv0 = fmaxf(base[c + 0] + pbv.x + pcv.x, 0.0f);
            float hv1 = fmaxf(base[c + 1] + pbv.y + pcv.y, 0.0f);
            float hv2 = fmaxf(base[c + 2] + pbv.z + pcv.z, 0.0f);
            float hv3 = fmaxf(base[c + 3] + pbv.w + pcv.w, 0.0f);
            #pragma unroll
            for (int hh = 0; hh < 8; ++hh) {
                acc[hh] += hv0 * w2l[(c + 0) * 8 + hh];
                acc[hh] += hv1 * w2l[(c + 1) * 8 + hh];
                acc[hh] += hv2 * w2l[(c + 2) * 8 + hh];
                acc[hh] += hv3 * w2l[(c + 3) * 8 + hh];
            }
        }
        #pragma unroll
        for (int hh = 0; hh < 8; ++hh) {
            float s = acc[hh] + b2[hh];
            Aout[(((long)b * 8 + hh) * 100 + i) * 100 + j] = 1.0f / (1.0f + expf(-s));
        }
    }
}

// ---------------------------------------------------------------------------
// Small elementwise helpers
// ---------------------------------------------------------------------------
__global__ void concat_logits_kernel(const float* __restrict__ sl, const float* __restrict__ al,
                                     float* __restrict__ p, int n)
{
    int i = blockIdx.x * blockDim.x + threadIdx.x;
    if (i < n) {
        int r = i / 19, c = i % 19;
        p[i] = (c < 7) ? sl[r * 7 + c] : al[r * 12 + (c - 7)];
    }
}

__global__ void add_relu_inplace_kernel(float* __restrict__ x, const float* __restrict__ u, int n4)
{
    int i = blockIdx.x * blockDim.x + threadIdx.x;
    if (i < n4) {
        float4 xv = ((const float4*)x)[i];
        float4 uv = ((const float4*)u)[i];
        xv.x += fmaxf(uv.x, 0.0f);
        xv.y += fmaxf(uv.y, 0.0f);
        xv.z += fmaxf(uv.z, 0.0f);
        xv.w += fmaxf(uv.w, 0.0f);
        ((float4*)x)[i] = xv;
    }
}

__global__ void add_kernel(const float* __restrict__ a, const float* __restrict__ b,
                           float* __restrict__ o, int n4)
{
    int i = blockIdx.x * blockDim.x + threadIdx.x;
    if (i < n4) {
        float4 av = ((const float4*)a)[i];
        float4 bv = ((const float4*)b)[i];
        float4 ov = { av.x + bv.x, av.y + bv.y, av.z + bv.z, av.w + bv.w };
        ((float4*)o)[i] = ov;
    }
}

// ---------------------------------------------------------------------------
// Host orchestration
// ---------------------------------------------------------------------------
extern "C" void kernel_launch(void* const* d_in, const int* in_sizes, int n_in,
                              void* d_out, int out_size, void* d_ws, size_t ws_size,
                              hipStream_t stream)
{
    (void)in_sizes; (void)n_in; (void)out_size; (void)ws_size;
    const float* sent_x      = (const float*)d_in[0];
    const float* act_x       = (const float*)d_in[1];
    const float* sent_logits = (const float*)d_in[2];
    const float* act_logits  = (const float*)d_in[3];
    const int*   len_list    = (const int*)d_in[4];
    const int*   adj         = (const int*)d_in[5];
    const float* w_sent_t    = (const float*)d_in[6];
    const float* b_sent_t    = (const float*)d_in[7];
    const float* w_act_t     = (const float*)d_in[8];
    const float* b_act_t     = (const float*)d_in[9];
    const float* w_sent_f    = (const float*)d_in[10];
    const float* b_sent_f    = (const float*)d_in[11];
    const float* w_act_f     = (const float*)d_in[12];
    const float* b_act_f     = (const float*)d_in[13];
    const float* sgen_w1     = (const float*)d_in[14];
    const float* sgen_b1     = (const float*)d_in[15];
    const float* sgen_w2     = (const float*)d_in[16];
    const float* sgen_b2     = (const float*)d_in[17];
    const float* agen_w1     = (const float*)d_in[18];
    const float* agen_b1     = (const float*)d_in[19];
    const float* agen_w2     = (const float*)d_in[20];
    const float* agen_b2     = (const float*)d_in[21];
    const float* pos_table   = (const float*)d_in[22];
    const float* s_qkv_w     = (const float*)d_in[23];
    const float* s_qkv_b     = (const float*)d_in[24];
    const float* s_o_w       = (const float*)d_in[25];
    const float* s_o_b       = (const float*)d_in[26];
    const float* a_qkv_w     = (const float*)d_in[27];
    const float* a_qkv_b     = (const float*)d_in[28];
    const float* a_o_w       = (const float*)d_in[29];
    const float* a_o_b       = (const float*)d_in[30];

    float* ws = (float*)d_ws;
    size_t off = 0;
    auto alloc = [&](size_t n) -> float* { float* r = ws + off; off += n; return r; };
    const size_t RH = 800 * 512;
    float* sx0  = alloc(RH);
    float* ax0  = alloc(RH);
    float* sx   = alloc(RH);
    float* ax   = alloc(RH);
    float* pcat = alloc(800 * 20);   // 19 used, padded
    float* pa_s = alloc(RH);  float* pb_s = alloc(RH);
    float* pa_a = alloc(RH);  float* pb_a = alloc(RH);
    float* pc_s = alloc(200 * 512);  float* pc_a = alloc(200 * 512);
    float* A_s  = alloc(8 * 8 * 100 * 100);
    float* A_a  = alloc(8 * 8 * 100 * 100);
    float* qb   = alloc(RH);  float* kb = alloc(RH);  float* vb = alloc(RH);
    float* ctx  = alloc(RH);  float* upd = alloc(RH); float* tmp = alloc(RH);

    auto gemm = [&](const float* A, const float* B, const float* bias, float* C,
                    int M, int N, int K, int lda, int ldb, int ldc, int accum) {
        dim3 grid((M + 31) / 32, (N + 31) / 32);
        if ((M % 32 == 0) && (N % 32 == 0) && (K % 32 == 0)) {
            wmma_gemm_kernel<false><<<grid, dim3(128), 0, stream>>>(A, B, bias, C,
                                                                    M, N, K, lda, ldb, ldc, accum);
        } else {
            wmma_gemm_kernel<true><<<grid, dim3(128), 0, stream>>>(A, B, bias, C,
                                                                   M, N, K, lda, ldb, ldc, accum);
        }
    };

    // Input transforms: sx0 = sent_x @ W + b, ax0 likewise; sx/ax start as copies.
    gemm(sent_x, w_sent_t, b_sent_t, sx0, 800, 512, 768, 768, 512, 512, 0);
    gemm(act_x,  w_act_t,  b_act_t,  ax0, 800, 512, 768, 768, 512, 512, 0);
    hipMemcpyAsync(sx, sx0, RH * sizeof(float), hipMemcpyDeviceToDevice, stream);
    hipMemcpyAsync(ax, ax0, RH * sizeof(float), hipMemcpyDeviceToDevice, stream);

    // Edge MLP decomposition: em@w1 = pa[i] + pb[j] + pc[j-i+100]
    concat_logits_kernel<<<(800 * 19 + 255) / 256, 256, 0, stream>>>(sent_logits, act_logits,
                                                                     pcat, 800 * 19);
    gemm(pcat, sgen_w1,               nullptr, pa_s, 800, 512, 19,  19,  512, 512, 0);
    gemm(pcat, sgen_w1 + 19 * 512,    nullptr, pb_s, 800, 512, 19,  19,  512, 512, 0);
    gemm(pos_table, sgen_w1 + 38 * 512, nullptr, pc_s, 200, 512, 512, 512, 512, 512, 0);
    gemm(pcat, agen_w1,               nullptr, pa_a, 800, 512, 19,  19,  512, 512, 0);
    gemm(pcat, agen_w1 + 19 * 512,    nullptr, pb_a, 800, 512, 19,  19,  512, 512, 0);
    gemm(pos_table, agen_w1 + 38 * 512, nullptr, pc_a, 200, 512, 512, 512, 512, 512, 0);
    edge_gen_kernel<<<800, 128, 0, stream>>>(pa_s, pb_s, pc_s, sgen_b1, sgen_w2, sgen_b2, A_s);
    edge_gen_kernel<<<800, 128, 0, stream>>>(pa_a, pb_a, pc_a, agen_b1, agen_w2, agen_b2, A_a);

    // Attention layers
    for (int l = 0; l < 2; ++l) {
        for (int br = 0; br < 2; ++br) {
            float* x          = (br == 0) ? sx  : ax;
            const float* Am   = (br == 0) ? A_s : A_a;
            const float* qkvw = (br == 0) ? s_qkv_w : a_qkv_w;
            const float* qkvb = (br == 0) ? s_qkv_b : a_qkv_b;
            const float* ow   = (br == 0) ? s_o_w   : a_o_w;
            const float* ob   = (br == 0) ? s_o_b   : a_o_b;
            for (int m = 0; m < 2; ++m) {
                long wo = (long)((l * 2 + m) * 3) * 512 * 512;
                long bo = (long)((l * 2 + m) * 3) * 512;
                gemm(x, qkvw + wo,                 qkvb + bo,        qb, 800, 512, 512, 512, 512, 512, 0);
                gemm(x, qkvw + wo + 512 * 512,     qkvb + bo + 512,  kb, 800, 512, 512, 512, 512, 512, 0);
                gemm(x, qkvw + wo + 2 * 512 * 512, qkvb + bo + 1024, vb, 800, 512, 512, 512, 512, 512, 0);
                attn_fused_kernel<<<64, 256, 0, stream>>>(qb, kb, vb, Am, adj, len_list, ctx, m);
                gemm(ctx, ow + (long)(l * 2 + m) * 512 * 512, ob + (l * 2 + m) * 512,
                     upd, 800, 512, 512, 512, 512, 512, m);  // m==1 -> accumulate
            }
            add_relu_inplace_kernel<<<((int)(RH / 4) + 255) / 256, 256, 0, stream>>>(x, upd, (int)(RH / 4));
        }
    }

    // Final heads: (x0 + x) @ Wf + bf
    float* out = (float*)d_out;
    add_kernel<<<((int)(RH / 4) + 255) / 256, 256, 0, stream>>>(sx0, sx, tmp, (int)(RH / 4));
    gemm(tmp, w_sent_f, b_sent_f, out, 800, 7, 512, 512, 7, 7, 0);
    add_kernel<<<((int)(RH / 4) + 255) / 256, 256, 0, stream>>>(ax0, ax, tmp, (int)(RH / 4));
    gemm(tmp, w_act_f, b_act_f, out + 5600, 800, 12, 512, 512, 12, 12, 0);
}
